// MemoryAugmentedLayer_89670327205934
// MI455X (gfx1250) — matrix operand order repaired and checked by reading, not compiled, not run on hardware
//
#include <hip/hip_runtime.h>

typedef __attribute__((ext_vector_type(16))) __bf16 v16bf;
typedef __attribute__((ext_vector_type(8)))  __bf16 v8bf;
typedef __attribute__((ext_vector_type(4)))  __bf16 v4bf;
typedef __attribute__((ext_vector_type(8)))  float  v8f;

#define B_ROWS 32768
#define DDIM   256
#define MDIM   1024
#define KVDIM  128
#define LDT    40          // LDS tile stride in elements (80B, 16B-aligned, bank-spread)

__device__ __forceinline__ float elu_f(float x) {
    return x > 0.0f ? x : (__expf(x) - 1.0f);
}

// ---------------------------------------------------------------------------
// Fragment loaders: contiguous 16B-aligned runs -> ds_load_b128 pairs.
// A tile row-major [row][LDT]; B tile column-major [col][LDT].
// ---------------------------------------------------------------------------
__device__ __forceinline__ v16bf frag_a(const __bf16* sA, int lane) {
    const int row = lane & 15, hf = lane >> 4;
    const __bf16* p = sA + row * LDT + hf * 8;
    union { v16bf v; v8bf h[2]; } u;
    u.h[0] = *(const v8bf*)p;
    u.h[1] = *(const v8bf*)(p + 16);
    return u.v;
}
__device__ __forceinline__ v16bf frag_b(const __bf16* sB, int colOff, int lane) {
    const int col = lane & 15, hf = lane >> 4;
    const __bf16* p = sB + (colOff + col) * LDT + hf * 8;
    union { v16bf v; v8bf h[2]; } u;
    u.h[0] = *(const v8bf*)p;
    u.h[1] = *(const v8bf*)(p + 16);
    return u.v;
}

// ---------------------------------------------------------------------------
// Prep 1: elementwise f32 -> bf16. 4 elems/thread, vectorized.
// ---------------------------------------------------------------------------
__global__ __launch_bounds__(256)
void cvt_kernel(const float* __restrict__ src, __bf16* __restrict__ dst)
{
    const int i = blockIdx.x * 256 + threadIdx.x;
    const float4 f = ((const float4*)src)[i];
    v4bf o;
    o[0] = (__bf16)f.x; o[1] = (__bf16)f.y; o[2] = (__bf16)f.z; o[3] = (__bf16)f.w;
    ((v4bf*)dst)[i] = o;
}

// ---------------------------------------------------------------------------
// Prep 2: transpose + convert, f32 [R][C] -> bf16 [C][R]. 32x32 tiles.
// ---------------------------------------------------------------------------
__global__ __launch_bounds__(256)
void transpose_cvt_kernel(const float* __restrict__ src, __bf16* __restrict__ dst,
                          int R, int C)
{
    __shared__ float tile[32][33];
    const int tx = threadIdx.x & 31, ty = threadIdx.x >> 5;
    const int r0 = blockIdx.y * 32, c0 = blockIdx.x * 32;
#pragma unroll
    for (int i = 0; i < 4; ++i)
        tile[ty + i * 8][tx] = src[(size_t)(r0 + ty + i * 8) * C + c0 + tx];
    __syncthreads();
#pragma unroll
    for (int i = 0; i < 4; ++i)
        dst[(size_t)(c0 + ty + i * 8) * R + r0 + tx] = (__bf16)tile[tx][ty + i * 8];
}

// ---------------------------------------------------------------------------
// Unified TN GEMM: out[rows][n] = act(A[rows][kdim] @ Bt[n][kdim]^T + bias)
//   Block tile 128x128, BK=32, 256 thr (8 waves in 2x4 grid).
//   Wave tile 64x32 -> acc[4][2], 8 WMMAs per k-step.
//   Double-buffered LDS + register prefetch: one barrier per k-step,
//   global latency hidden behind the WMMA burst.
// grid = (rows/128, n/128)
// ---------------------------------------------------------------------------
template<typename OutT, bool ELU_ACT, bool WRITE_T>
__global__ __launch_bounds__(256)
void gemm_tn_kernel(const __bf16* __restrict__ A, int kdim,
                    const __bf16* __restrict__ Bt,
                    const float* __restrict__ bias,
                    OutT* __restrict__ out, int n,
                    __bf16* __restrict__ outT, int ldT)
{
    __shared__ __align__(16) __bf16 sA[2][128 * LDT];
    __shared__ __align__(16) __bf16 sB[2][128 * LDT];
    const int tid = threadIdx.x, wave = tid >> 5, lane = tid & 31;
    const int waveRow = wave >> 2, waveCol = wave & 3;
    const int rowBase = blockIdx.x * 128;
    const int colBase = blockIdx.y * 128;

    // staging: thread -> (row tid>>1, k-half (tid&1)*16), 32B per matrix
    const int sr = tid >> 1, sk = (tid & 1) * 16;
    const __bf16* gA = A + (size_t)(rowBase + sr) * kdim + sk;
    const __bf16* gB = Bt + (size_t)(colBase + sr) * kdim + sk;
    const int soff = sr * LDT + sk;

    v8f acc[4][2] = {};

    // prologue: chunk 0 into buffer 0
    v8bf pa0 = *(const v8bf*)gA, pa1 = *(const v8bf*)(gA + 8);
    v8bf pb0 = *(const v8bf*)gB, pb1 = *(const v8bf*)(gB + 8);
    gA += 32; gB += 32;
    *(v8bf*)(&sA[0][soff]) = pa0; *(v8bf*)(&sA[0][soff + 8]) = pa1;
    *(v8bf*)(&sB[0][soff]) = pb0; *(v8bf*)(&sB[0][soff + 8]) = pb1;
    __syncthreads();

    const int nk = kdim >> 5;
    for (int i = 0; i < nk; ++i) {
        const int cur = i & 1;
        if (i + 1 < nk) {   // register prefetch of next chunk (no LDS yet)
            pa0 = *(const v8bf*)gA; pa1 = *(const v8bf*)(gA + 8);
            pb0 = *(const v8bf*)gB; pb1 = *(const v8bf*)(gB + 8);
            gA += 32; gB += 32;
        }
        // hoist all fragment loads, then burst 8 WMMAs
        const __bf16* cA = sA[cur] + (waveRow * 64) * LDT;
        const __bf16* cB = sB[cur];
        const v16bf bf0 = frag_b(cB, waveCol * 32, lane);
        const v16bf bf1 = frag_b(cB, waveCol * 32 + 16, lane);
        v16bf af[4];
#pragma unroll
        for (int rt = 0; rt < 4; ++rt) af[rt] = frag_a(cA + rt * 16 * LDT, lane);
#pragma unroll
        for (int rt = 0; rt < 4; ++rt) {
            acc[rt][0] = __builtin_amdgcn_wmma_f32_16x16x32_bf16(
                false, af[rt], false, bf0, (short)0, acc[rt][0], false, false);
            acc[rt][1] = __builtin_amdgcn_wmma_f32_16x16x32_bf16(
                false, af[rt], false, bf1, (short)0, acc[rt][1], false, false);
        }
        if (i + 1 < nk) {   // write prefetched chunk to the other buffer
            const int nxt = cur ^ 1;
            *(v8bf*)(&sA[nxt][soff]) = pa0; *(v8bf*)(&sA[nxt][soff + 8]) = pa1;
            *(v8bf*)(&sB[nxt][soff]) = pb0; *(v8bf*)(&sB[nxt][soff + 8]) = pb1;
        }
        __syncthreads();
    }

    const int colL = lane & 15, hf = lane >> 4;
#pragma unroll
    for (int ct = 0; ct < 2; ++ct) {
        const int col = colBase + waveCol * 32 + ct * 16 + colL;
        const float bv = bias ? bias[col] : 0.0f;
#pragma unroll
        for (int rt = 0; rt < 4; ++rt) {
#pragma unroll
            for (int v = 0; v < 8; ++v) {
                const int row = rowBase + waveRow * 64 + rt * 16 + hf * 8 + v;
                float val = acc[rt][ct][v] + bv;
                if (ELU_ACT) val = elu_f(val);
                out[(size_t)row * n + col] = (OutT)val;
                if (WRITE_T) outT[(size_t)col * ldT + row] = (__bf16)val;
            }
        }
    }
}

// ---------------------------------------------------------------------------
// Row softmax: f32 [B,1024] -> bf16 [B,1024]. One block / row, vectorized.
// ---------------------------------------------------------------------------
__global__ __launch_bounds__(256)
void softmax_rows_kernel(const float* __restrict__ logits, __bf16* __restrict__ out)
{
    __shared__ float red[8];
    const int row = blockIdx.x, tid = threadIdx.x;
    const int wave = tid >> 5, lane = tid & 31;
    const float4 f = ((const float4*)(logits + (size_t)row * MDIM))[tid];
    float x[4] = { f.x, f.y, f.z, f.w };
    float mx = fmaxf(fmaxf(x[0], x[1]), fmaxf(x[2], x[3]));
#pragma unroll
    for (int off = 16; off; off >>= 1) mx = fmaxf(mx, __shfl_xor(mx, off, 32));
    if (lane == 0) red[wave] = mx;
    __syncthreads();
    mx = red[0];
#pragma unroll
    for (int i = 1; i < 8; ++i) mx = fmaxf(mx, red[i]);
    __syncthreads();
    float s = 0.0f;
#pragma unroll
    for (int i = 0; i < 4; ++i) { x[i] = __expf(x[i] - mx); s += x[i]; }
#pragma unroll
    for (int off = 16; off; off >>= 1) s += __shfl_xor(s, off, 32);
    if (lane == 0) red[wave] = s;
    __syncthreads();
    s = 0.0f;
#pragma unroll
    for (int i = 0; i < 8; ++i) s += red[i];
    const float inv = 1.0f / s;
    v4bf o;
#pragma unroll
    for (int i = 0; i < 4; ++i) o[i] = (__bf16)(x[i] * inv);
    ((v4bf*)(out + (size_t)row * MDIM))[tid] = o;
}

// ---------------------------------------------------------------------------
// Memory update: mem_bf16 = bf16( memOld_f32 + (probs^T @ vec) / B )
//   probs: bf16 [B,1024]; vecT: bf16 [128][B]. Block 64(m) x 128(cols),
//   grid 16, double-buffered LDS + register prefetch over the batch dim.
// ---------------------------------------------------------------------------
template<bool TOUT>
__global__ __launch_bounds__(256)
void mem_update_kernel(const __bf16* __restrict__ probs,
                       const __bf16* __restrict__ vecT,
                       const float* __restrict__ memOld,
                       __bf16* __restrict__ outMem)
{
    __shared__ __align__(16) __bf16 sA[2][64 * LDT];
    __shared__ __align__(16) __bf16 sB[2][128 * LDT];
    const int tid = threadIdx.x, wave = tid >> 5, lane = tid & 31;
    const int mBase = blockIdx.x * 64;

    // A^T staging: thread -> (batch row tid>>3, 8 m's at (tid&7)*8)
    const int abb = tid >> 3, ams = (tid & 7) * 8;
    const __bf16* gA = probs + (size_t)abb * MDIM + mBase + ams;
    // B staging: thread -> (col tid>>1, k-half (tid&1)*16) from vecT
    const int bcc = tid >> 1, bks = (tid & 1) * 16;
    const __bf16* gB = vecT + (size_t)bcc * B_ROWS + bks;
    const int boff = bcc * LDT + bks;

    v8f acc[4] = {};

    v8bf pa = *(const v8bf*)gA;
    v8bf pb0 = *(const v8bf*)gB, pb1 = *(const v8bf*)(gB + 8);
    gA += (size_t)32 * MDIM; gB += 32;
#pragma unroll
    for (int j = 0; j < 8; ++j) sA[0][(ams + j) * LDT + abb] = pa[j];
    *(v8bf*)(&sB[0][boff]) = pb0; *(v8bf*)(&sB[0][boff + 8]) = pb1;
    __syncthreads();

    const int nk = B_ROWS / 32;
    for (int i = 0; i < nk; ++i) {
        const int cur = i & 1;
        if (i + 1 < nk) {
            pa = *(const v8bf*)gA;
            pb0 = *(const v8bf*)gB; pb1 = *(const v8bf*)(gB + 8);
            gA += (size_t)32 * MDIM; gB += 32;
        }
        const v16bf b = frag_b(sB[cur], wave * 16, lane);
        v16bf af[4];
#pragma unroll
        for (int rt = 0; rt < 4; ++rt) af[rt] = frag_a(sA[cur] + rt * 16 * LDT, lane);
#pragma unroll
        for (int rt = 0; rt < 4; ++rt)
            acc[rt] = __builtin_amdgcn_wmma_f32_16x16x32_bf16(
                false, af[rt], false, b, (short)0, acc[rt], false, false);
        if (i + 1 < nk) {
            const int nxt = cur ^ 1;
#pragma unroll
            for (int j = 0; j < 8; ++j) sA[nxt][(ams + j) * LDT + abb] = pa[j];
            *(v8bf*)(&sB[nxt][boff]) = pb0; *(v8bf*)(&sB[nxt][boff + 8]) = pb1;
        }
        __syncthreads();
    }

    const float invB = 1.0f / (float)B_ROWS;
    const int colL = lane & 15, hf = lane >> 4;
    const int col = wave * 16 + colL;
#pragma unroll
    for (int rt = 0; rt < 4; ++rt) {
#pragma unroll
        for (int v = 0; v < 8; ++v) {
            const int m = mBase + rt * 16 + hf * 8 + v;
            const float val = memOld[(size_t)m * KVDIM + col] + acc[rt][v] * invB;
            if (TOUT) outMem[(size_t)col * MDIM + m] = (__bf16)val;
            else      outMem[(size_t)m * KVDIM + col] = (__bf16)val;
        }
    }
}

// ---------------------------------------------------------------------------
extern "C" void kernel_launch(void* const* d_in, const int* in_sizes, int n_in,
                              void* d_out, int out_size, void* d_ws, size_t ws_size,
                              hipStream_t stream)
{
    const float* x      = (const float*)d_in[0];
    const float* Wk     = (const float*)d_in[1];
    const float* bk     = (const float*)d_in[2];
    const float* Wv     = (const float*)d_in[3];
    const float* bv     = (const float*)d_in[4];
    const float* Wq     = (const float*)d_in[5];
    const float* bq     = (const float*)d_in[6];
    const float* Wwr    = (const float*)d_in[7];
    const float* bwr    = (const float*)d_in[8];
    const float* Wrd    = (const float*)d_in[9];
    const float* brd    = (const float*)d_in[10];
    const float* keyMem = (const float*)d_in[11];
    const float* valMem = (const float*)d_in[12];
    float* out = (float*)d_out;

    const size_t Bx = B_ROWS, Mx = MDIM;
    __bf16* p = (__bf16*)d_ws;
    __bf16* xBf    = p; p += Bx * DDIM;       // [B,256]
    __bf16* keyV   = p; p += Bx * KVDIM;      // [B,128]
    __bf16* valV   = p; p += Bx * KVDIM;
    __bf16* qryV   = p; p += Bx * KVDIM;
    __bf16* keyVT  = p; p += Bx * KVDIM;      // [128,B]
    __bf16* valVT  = p; p += Bx * KVDIM;
    __bf16* scoreB = p; p += Bx * Mx;         // [B,1024]
    __bf16* probB  = p; p += Bx * Mx;         // [B,1024]
    __bf16* WkT    = p; p += DDIM * KVDIM;    // [128,256]
    __bf16* WvT    = p; p += DDIM * KVDIM;
    __bf16* WqT    = p; p += DDIM * KVDIM;
    __bf16* WwrT   = p; p += Mx * Mx;         // [1024,1024]
    __bf16* WrdT   = p; p += Mx * Mx;
    __bf16* keyMB  = p; p += Mx * KVDIM;      // [1024,128]
    __bf16* kmemN  = p; p += Mx * KVDIM;      // [1024,128]
    __bf16* vmemT  = p; p += Mx * KVDIM;      // [128,1024]
    float*  logits = (float*)p;               // [B,1024] f32

    const dim3 blk(256);
    const dim3 gRows(B_ROWS / 128);                 // 256
    const dim3 gScore(B_ROWS / 128, MDIM / 128);    // 256 x 8

    // --- prep: bf16 conversions and transposed-bf16 weights ---
    cvt_kernel<<<dim3((Bx * DDIM) / 1024), blk, 0, stream>>>(x, xBf);
    cvt_kernel<<<dim3((Mx * KVDIM) / 1024), blk, 0, stream>>>(keyMem, keyMB);
    transpose_cvt_kernel<<<dim3(KVDIM / 32, DDIM / 32), blk, 0, stream>>>(Wk, WkT, DDIM, KVDIM);
    transpose_cvt_kernel<<<dim3(KVDIM / 32, DDIM / 32), blk, 0, stream>>>(Wv, WvT, DDIM, KVDIM);
    transpose_cvt_kernel<<<dim3(KVDIM / 32, DDIM / 32), blk, 0, stream>>>(Wq, WqT, DDIM, KVDIM);
    transpose_cvt_kernel<<<dim3(MDIM / 32, MDIM / 32), blk, 0, stream>>>(Wwr, WwrT, MDIM, MDIM);
    transpose_cvt_kernel<<<dim3(MDIM / 32, MDIM / 32), blk, 0, stream>>>(Wrd, WrdT, MDIM, MDIM);

    // --- projections + ELU (key/val also emit transposed copies) ---
    gemm_tn_kernel<__bf16, true, true ><<<gRows, blk, 0, stream>>>(xBf, DDIM, WkT, bk, keyV, KVDIM, keyVT, B_ROWS);
    gemm_tn_kernel<__bf16, true, true ><<<gRows, blk, 0, stream>>>(xBf, DDIM, WvT, bv, valV, KVDIM, valVT, B_ROWS);
    gemm_tn_kernel<__bf16, true, false><<<gRows, blk, 0, stream>>>(xBf, DDIM, WqT, bq, qryV, KVDIM, nullptr, 0);

    // --- write path ---
    gemm_tn_kernel<__bf16, false, false><<<gScore, blk, 0, stream>>>(keyV, KVDIM, keyMB, nullptr, scoreB, MDIM, nullptr, 0);
    gemm_tn_kernel<float,  false, false><<<gScore, blk, 0, stream>>>(scoreB, MDIM, WwrT, bwr, logits, MDIM, nullptr, 0);
    softmax_rows_kernel<<<dim3(B_ROWS), blk, 0, stream>>>(logits, probB);

    // --- memory updates (used by the read path) ---
    mem_update_kernel<false><<<dim3(MDIM / 64), blk, 0, stream>>>(probB, keyVT, keyMem, kmemN);
    mem_update_kernel<true ><<<dim3(MDIM / 64), blk, 0, stream>>>(probB, valVT, valMem, vmemT);

    // --- read path ---
    gemm_tn_kernel<__bf16, false, false><<<gScore, blk, 0, stream>>>(qryV, KVDIM, kmemN, nullptr, scoreB, MDIM, nullptr, 0);
    gemm_tn_kernel<float,  false, false><<<gScore, blk, 0, stream>>>(scoreB, MDIM, WrdT, brd, logits, MDIM, nullptr, 0);
    softmax_rows_kernel<<<dim3(B_ROWS), blk, 0, stream>>>(logits, probB);

    // --- read_vec = read_w @ val_mem_new -> d_out (f32 [B,128]) ---
    gemm_tn_kernel<float, false, false><<<dim3(B_ROWS / 128, 1), blk, 0, stream>>>(probB, MDIM, vmemT, nullptr, out, KVDIM, nullptr, 0);
}
